// SNNNetwork_82875688944065
// MI455X (gfx1250) — matrix operand order repaired and checked by reading, not compiled, not run on hardware
//
#include <hip/hip_runtime.h>

typedef __attribute__((ext_vector_type(16))) _Float16 v16h;
typedef __attribute__((ext_vector_type(8)))  float    v8f;
typedef __attribute__((ext_vector_type(2)))  float    v2f;

#define B_   256
#define T_   1024
#define S_   7
#define H_   64
#define NL   4
#define OUT_ 3

#if defined(__has_builtin)
#if __has_builtin(__builtin_amdgcn_wmma_f32_16x16x4_f32)
#define HAS_WMMA_F32X4 1
#endif
#endif

__launch_bounds__(128, 1)
__global__ void snn_scan_kernel(const float* __restrict__ x,
                                const float* __restrict__ hs0,
                                const float* __restrict__ w1,
                                const float* __restrict__ b1,
                                const float* __restrict__ w_h,
                                const float* __restrict__ b_h,
                                const float* __restrict__ w_out,
                                const float* __restrict__ b_out,
                                float* __restrict__ out_y,    // (B,T,3)
                                float* __restrict__ out_hid)  // (B,T,4,64)
{
    // Spike A-operand image, double buffered, laid out as 16-bit A 16x32 (k = 16*(lane>>4) + s)
    // register image: [buf][kblock(32)][lane][16 halves].
    __shared__ __align__(64) _Float16 spkimg[2][2][32][16];

    const int lane  = threadIdx.x & 31;
    const int wave  = threadIdx.x >> 5;
    const int jloc  = lane & 15;       // N within tile
    const int khalf = lane >> 4;       // lane half
    const int jglob = wave * 16 + jloc; // hidden unit this wave's tile owns
    const int b0    = blockIdx.x * 16;

    // ---- biases ----
    float bias[NL];
    bias[0] = b1[jglob];
#pragma unroll
    for (int l = 1; l < NL; ++l) bias[l] = b_h[(l - 1) * H_ + jglob];
    const float biasOut = (jloc < OUT_) ? b_out[jloc] : 0.0f;

    // ---- w1 as f32 WMMA B-operand (4x16, two K-steps, K=7 padded) ----
    const float* w1r = w1 + jglob * S_;
    v2f bw0, bw1;
    bw0.x = w1r[2 * khalf];                       // K = 2*half
    bw0.y = w1r[2 * khalf + 1];                   // K = 2*half+1
    bw1.x = w1r[4 + 2 * khalf];                   // K = 4/6
    bw1.y = (khalf == 0) ? w1r[5] : 0.0f;         // K = 5 / 7(pad)
#ifndef HAS_WMMA_F32X4
    float w1v[S_];
#pragma unroll
    for (int s = 0; s < S_; ++s) w1v[s] = w1r[s];
#endif

    // ---- hidden weights, split f32 -> f16 hi + f16 lo, resident in VGPRs ----
    // B 32x16 f16 order used here (contiguous k): lane L, half h -> K = h + 16*(L>>4), N = L&15.
    v16h whi[NL - 1][2], wlo[NL - 1][2];
#pragma unroll
    for (int l = 0; l < NL - 1; ++l) {
#pragma unroll
        for (int kb = 0; kb < 2; ++kb) {
            const float* wr = w_h + ((size_t)l * H_ + jglob) * H_ + kb * 32 + 16 * khalf;
#pragma unroll
            for (int h = 0; h < 16; ++h) {
                float w = wr[h];
                _Float16 hi = (_Float16)w;
                whi[l][kb][h] = hi;
                wlo[l][kb][h] = (_Float16)(w - (float)hi);
            }
        }
    }
    // output head weights (3x64 padded to 16 cols); only wave 0 consumes
    v16h wohi[2], wolo[2];
#pragma unroll
    for (int kb = 0; kb < 2; ++kb) {
        const float* wr = w_out + (size_t)jloc * H_ + kb * 32 + 16 * khalf;
#pragma unroll
        for (int h = 0; h < 16; ++h) {
            float w = (jloc < OUT_) ? wr[h] : 0.0f;
            _Float16 hi = (_Float16)w;
            wohi[kb][h] = hi;
            wolo[kb][h] = (_Float16)(w - (float)hi);
        }
    }

    // spike export coordinates (A-image slot for hidden index kg = this lane's col)
    const int kg  = wave * 16 + jloc;
    const int skb = kg >> 5;
    const int skk = kg & 31;
    const int slh = skk >> 4;                        // target lane half
    const int sh  = skk & 15;                        // half slot within v16h

    // ---- membrane state in registers: 4 layers x 8 accumulator rows ----
    float memr[NL][8];
#pragma unroll
    for (int l = 0; l < NL; ++l)
#pragma unroll
        for (int r = 0; r < 8; ++r) {
            unsigned b = b0 + r + 8 * khalf;
            memr[l][r] = hs0[((size_t)b * T_ * NL + l) * H_ + jglob];
        }

    const float* xrow = x + (size_t)(b0 + jloc) * T_ * S_;  // A rows: M = lane&15
    unsigned hidbase[8];
#pragma unroll
    for (int r = 0; r < 8; ++r)
        hidbase[r] = (unsigned)(b0 + r + 8 * khalf) * (unsigned)(T_ * NL * H_) + (unsigned)jglob;

    for (int t = 0; t < T_; ++t) {
        // ---------- stage 0: layer-0 current from x ----------
        const float* xp = xrow + (size_t)t * S_;
        if (t + 8 < T_) __builtin_prefetch(xrow + (size_t)(t + 8) * S_, 0, 1);

        v8f acc;
#pragma unroll
        for (int i = 0; i < 8; ++i) acc[i] = bias[0];
#ifdef HAS_WMMA_F32X4
        {
            v2f a0, a1;
            a0.x = xp[2 * khalf];
            a0.y = xp[2 * khalf + 1];
            a1.x = xp[4 + 2 * khalf];
            a1.y = (khalf == 0) ? xp[5] : 0.0f;
            acc = __builtin_amdgcn_wmma_f32_16x16x4_f32(false, a0, false, bw0,
                                                        (short)0, acc, false, false);
            acc = __builtin_amdgcn_wmma_f32_16x16x4_f32(false, a1, false, bw1,
                                                        (short)0, acc, false, false);
        }
#else
#pragma unroll
        for (int r = 0; r < 8; ++r) {
            const float* xm = x + ((size_t)(b0 + r + 8 * khalf) * T_ + t) * S_;
            float s = 0.0f;
#pragma unroll
            for (int si = 0; si < S_; ++si) s = fmaf(xm[si], w1v[si], s);
            acc[r] += s;
        }
#endif

        // membrane update layer 0 + spike export to buf 0
#pragma unroll
        for (int r = 0; r < 8; ++r) {
            float mp = memr[0][r];
            float reset = (mp > 1.0f) ? 1.0f : 0.0f;
            float mn = fmaf(0.8f, mp, acc[r]) - reset;
            memr[0][r] = mn;
            out_hid[hidbase[r] + (unsigned)t * (NL * H_)] = mn;
            spkimg[0][skb][(r + 8 * khalf) + 16 * slh][sh] =
                (mn - 1.0f > 0.0f) ? (_Float16)1.0f : (_Float16)0.0f;
        }
        __syncthreads();

        // ---------- stages 1..3: hidden layers via f16-split WMMA ----------
#pragma unroll
        for (int l = 1; l < NL; ++l) {
            const int cb = (l - 1) & 1;
            v16h A0 = *(const v16h*)&spkimg[cb][0][lane][0];
            v16h A1 = *(const v16h*)&spkimg[cb][1][lane][0];
#pragma unroll
            for (int i = 0; i < 8; ++i) acc[i] = bias[l];
            acc = __builtin_amdgcn_wmma_f32_16x16x32_f16(false, A0, false, whi[l - 1][0],
                                                         (short)0, acc, false, false);
            acc = __builtin_amdgcn_wmma_f32_16x16x32_f16(false, A0, false, wlo[l - 1][0],
                                                         (short)0, acc, false, false);
            acc = __builtin_amdgcn_wmma_f32_16x16x32_f16(false, A1, false, whi[l - 1][1],
                                                         (short)0, acc, false, false);
            acc = __builtin_amdgcn_wmma_f32_16x16x32_f16(false, A1, false, wlo[l - 1][1],
                                                         (short)0, acc, false, false);

            const int pb = l & 1;
#pragma unroll
            for (int r = 0; r < 8; ++r) {
                float mp = memr[l][r];
                float reset = (mp > 1.0f) ? 1.0f : 0.0f;
                float mn = fmaf(0.8f, mp, acc[r]) - reset;
                memr[l][r] = mn;
                out_hid[hidbase[r] + (unsigned)t * (NL * H_) + (unsigned)(l * H_)] = mn;
                spkimg[pb][skb][(r + 8 * khalf) + 16 * slh][sh] =
                    (mn - 1.0f > 0.0f) ? (_Float16)1.0f : (_Float16)0.0f;
            }
            __syncthreads();
        }

        // ---------- output head: wave 0 only (reads layer-3 spikes, buf 1) ----------
        if (wave == 0) {
            v16h A0 = *(const v16h*)&spkimg[1][0][lane][0];
            v16h A1 = *(const v16h*)&spkimg[1][1][lane][0];
            v8f oacc;
#pragma unroll
            for (int i = 0; i < 8; ++i) oacc[i] = biasOut;
            oacc = __builtin_amdgcn_wmma_f32_16x16x32_f16(false, A0, false, wohi[0],
                                                          (short)0, oacc, false, false);
            oacc = __builtin_amdgcn_wmma_f32_16x16x32_f16(false, A0, false, wolo[0],
                                                          (short)0, oacc, false, false);
            oacc = __builtin_amdgcn_wmma_f32_16x16x32_f16(false, A1, false, wohi[1],
                                                          (short)0, oacc, false, false);
            oacc = __builtin_amdgcn_wmma_f32_16x16x32_f16(false, A1, false, wolo[1],
                                                          (short)0, oacc, false, false);
            if (jloc < OUT_) {
#pragma unroll
                for (int r = 0; r < 8; ++r) {
                    out_y[((unsigned)(b0 + r + 8 * khalf) * T_ + (unsigned)t) * OUT_ + jloc] =
                        oacc[r];
                }
            }
        }
        // no barrier needed: next stage-0 writes buf 0; wave 0 reads only buf 1.
    }
}

__global__ void copy_x4_kernel(const float4* __restrict__ src,
                               float4* __restrict__ dst, int n) {
    int i = blockIdx.x * blockDim.x + threadIdx.x;
    int stride = gridDim.x * blockDim.x;
    for (; i < n; i += stride) dst[i] = src[i];
}

extern "C" void kernel_launch(void* const* d_in, const int* in_sizes, int n_in,
                              void* d_out, int out_size, void* d_ws, size_t ws_size,
                              hipStream_t stream) {
    (void)in_sizes; (void)n_in; (void)out_size; (void)d_ws; (void)ws_size;
    const float* x     = (const float*)d_in[0];
    const float* hs    = (const float*)d_in[1];
    /* d_in[2] = prev_obs (unused by reference math) */
    const float* w1    = (const float*)d_in[3];
    const float* b1    = (const float*)d_in[4];
    const float* w_h   = (const float*)d_in[5];
    const float* b_h   = (const float*)d_in[6];
    const float* w_out = (const float*)d_in[7];
    const float* b_out = (const float*)d_in[8];

    float* out_y   = (float*)d_out;                                  // B*T*3
    float* out_hid = out_y + (size_t)B_ * T_ * OUT_;                 // B*T*4*64
    float* out_x   = out_hid + (size_t)B_ * T_ * NL * H_;            // B*T*7

    snn_scan_kernel<<<B_ / 16, 128, 0, stream>>>(x, hs, w1, b1, w_h, b_h,
                                                 w_out, b_out, out_y, out_hid);

    int n4 = (B_ * T_ * S_) / 4;
    int blocks = (n4 + 255) / 256;
    copy_x4_kernel<<<blocks, 256, 0, stream>>>((const float4*)x, (float4*)out_x, n4);
}
